// Net_21603685499163
// MI455X (gfx1250) — compile-verified, matrix operand order checked by reading, not run on hardware
//
#include <hip/hip_runtime.h>
#include <hip/hip_bf16.h>
#include <stdint.h>

// ---------------- problem constants ----------------
#define NN    10000      // nodes
#define EE    200000     // edges
#define F_IN  512
#define CC    128
#define HC    256        // H*C
#define NC1   896        // q(256)|k(256)|v(256)|skip(128)
#define SCALE 0.08838834764831845f   // 1/sqrt(128)

typedef __attribute__((ext_vector_type(16))) __bf16 v16bf;
typedef __attribute__((ext_vector_type(8)))  float  v8f;
typedef __attribute__((ext_vector_type(4)))  float  f4v;   // native vector for NT builtins

static __device__ __forceinline__ unsigned short f2bf(float f) {
  unsigned u = __float_as_uint(f);
  unsigned r = u + 0x7FFFu + ((u >> 16) & 1u);   // round-to-nearest-even
  return (unsigned short)(r >> 16);
}
// order-preserving float<->uint encode for atomicMax-based segment max
static __device__ __forceinline__ unsigned fenc(float f) {
  unsigned b = __float_as_uint(f);
  return (b & 0x80000000u) ? ~b : (b | 0x80000000u);
}
static __device__ __forceinline__ float fdec(unsigned k) {
  unsigned b = (k & 0x80000000u) ? (k & 0x7FFFFFFFu) : ~k;
  return __uint_as_float(b);
}

// ---------------- generic fills / convert ----------------
__global__ void k_fill_f32(float* __restrict__ p, float v, int n) {
  int t = blockIdx.x * blockDim.x + threadIdx.x;
  if (t < n) p[t] = v;
}
__global__ void k_fill_u32(unsigned* __restrict__ p, unsigned v, int n) {
  int t = blockIdx.x * blockDim.x + threadIdx.x;
  if (t < n) p[t] = v;
}
__global__ void k_cvt_bf16(const float* __restrict__ src, unsigned short* __restrict__ dst, int n) {
  int t = blockIdx.x * blockDim.x + threadIdx.x;
  if (t < n) dst[t] = f2bf(src[t]);
}

// ---------------- swizzled weight packing ----------------
// Output layout = WMMA B-fragment order:
//   flat = (((nt*kTiles + kt)*32 + lane)*16 + i)
//   col  j = nt*16 + lane%16
//   row  k = kt*32 + i + (i&8) + ((lane>>4)<<3)    (ISA 7.12.2 16-bit B K-pair map)
// Each lane's 16 bf16 are contiguous -> B fragment = two coalesced b128 loads.
__global__ void k_pack_wcat_swz(const float* __restrict__ Wq, const float* __restrict__ Wk,
                                const float* __restrict__ Wv, const float* __restrict__ Ws,
                                unsigned short* __restrict__ out, int K) {
  int t = blockIdx.x * blockDim.x + threadIdx.x;
  if (t >= K * NC1) return;
  int i = t & 15, lane = (t >> 4) & 31, rest = t >> 9;
  int kTiles = K >> 5;
  int kt = rest % kTiles, nt = rest / kTiles;
  int j = nt * 16 + (lane & 15);
  int k = kt * 32 + i + (i & 8) + ((lane >> 4) << 3);
  float v;
  if (j < 256)      v = Wq[k * 256 + j];
  else if (j < 512) v = Wk[k * 256 + (j - 256)];
  else if (j < 768) v = Wv[k * 256 + (j - 512)];
  else              v = Ws[k * 128 + (j - 768)];
  out[t] = f2bf(v);
}
// Wp1 [256,128] -> swizzled bf16 [128,256]: cols 0..127 src-part rows, 128..255 dst-part rows
__global__ void k_pack_wpcat_swz(const float* __restrict__ Wp1, unsigned short* __restrict__ out) {
  int t = blockIdx.x * blockDim.x + threadIdx.x;
  if (t >= 128 * 256) return;
  int i = t & 15, lane = (t >> 4) & 31, rest = t >> 9;
  int kTiles = 128 >> 5;                         // 4
  int kt = rest % kTiles, nt = rest / kTiles;
  int j = nt * 16 + (lane & 15);
  int k = kt * 32 + i + (i & 8) + ((lane >> 4) << 3);
  float v = (j < 128) ? Wp1[k * 128 + j] : Wp1[(k + 128) * 128 + (j - 128)];
  out[t] = f2bf(v);
}
__global__ void k_pack_bcat(const float* __restrict__ bq, const float* __restrict__ bk,
                            const float* __restrict__ bv, const float* __restrict__ bs,
                            float* __restrict__ out) {
  int j = blockIdx.x * blockDim.x + threadIdx.x;
  if (j >= NC1) return;
  float v;
  if (j < 256)      v = bq[j];
  else if (j < 512) v = bk[j - 256];
  else if (j < 768) v = bv[j - 512];
  else              v = bs[j - 768];
  out[j] = v;
}
// We2eff[6,256] = mean-head(We1) @ We2   (folds layer-1 edge update into layer-2 edge proj)
__global__ void k_we2eff(const float* __restrict__ We1, const float* __restrict__ We2,
                         float* __restrict__ out) {
  int t = blockIdx.x * blockDim.x + threadIdx.x;
  if (t >= 6 * 256) return;
  int d = t >> 8, j = t & 255;
  float acc = 0.f;
  for (int c = 0; c < 128; ++c)
    acc += 0.5f * (We1[d * 256 + c] + We1[d * 256 + 128 + c]) * We2[c * 256 + j];
  out[t] = acc;
}

// ---------------- bf16 WMMA GEMM: C[M,Nc] = A[M,K] @ Bswz[K,Nc] (+bias) ----------------
// One wave -> 16 rows x 64 cols (4 accumulators, 4 v_wmma per K step).
// A fragment: two b128 loads at byte offsets 16*hi / 32+16*hi within the 64B K-slab.
// B fragment: two b128 loads from the pre-swizzled weight (lane-contiguous 32B).
#define WMMA_BF16(A_, B_, C_) \
  __builtin_amdgcn_wmma_f32_16x16x32_bf16(false, (A_), false, (B_), (short)0, (C_), false, false)

__global__ void k_gemm_bf16(const unsigned short* __restrict__ A, const unsigned short* __restrict__ B,
                            const float* __restrict__ bias, float* __restrict__ C,
                            int M, int K, int Nc) {
  const int lane   = threadIdx.x & 31;
  const int wave   = threadIdx.x >> 5;
  const int ntBase = blockIdx.x * 4;                 // 4 x 16 output cols per block
  const int mBase  = blockIdx.y * 64 + wave * 16;
  if (mBase >= M) return;                            // wave-uniform: EXEC all-ones at WMMA
  const int l16 = lane & 15;
  const int hi  = lane >> 4;
  const int kTiles = K >> 5;
  const unsigned short* Arow = A + (size_t)(mBase + l16) * K;
  const unsigned short* Bb   = B + ((size_t)ntBase * kTiles * 512) + (size_t)lane * 16;
  const size_t bStride = (size_t)kTiles * 512;       // elems between 16-col subtiles
  v8f acc0 = {}, acc1 = {}, acc2 = {}, acc3 = {};
  for (int kt = 0; kt < kTiles; ++kt) {
    const int k0 = kt << 5;
    union { uint4 q[2]; v16bf v; } af, b0, b1, b2, b3;
    af.q[0] = *(const uint4*)(Arow + k0 + hi * 8);
    af.q[1] = *(const uint4*)(Arow + k0 + 16 + hi * 8);
    __builtin_prefetch(Arow + k0 + 64, 0, 1);        // next-but-one K slab
    const unsigned short* bp = Bb + (size_t)kt * 512;
    b0.q[0] = *(const uint4*)(bp);
    b0.q[1] = *(const uint4*)(bp + 8);
    b1.q[0] = *(const uint4*)(bp + bStride);
    b1.q[1] = *(const uint4*)(bp + bStride + 8);
    b2.q[0] = *(const uint4*)(bp + 2 * bStride);
    b2.q[1] = *(const uint4*)(bp + 2 * bStride + 8);
    b3.q[0] = *(const uint4*)(bp + 3 * bStride);
    b3.q[1] = *(const uint4*)(bp + 3 * bStride + 8);
    acc0 = WMMA_BF16(af.v, b0.v, acc0);
    acc1 = WMMA_BF16(af.v, b1.v, acc1);
    acc2 = WMMA_BF16(af.v, b2.v, acc2);
    acc3 = WMMA_BF16(af.v, b3.v, acc3);
  }
  // C/D layout: M = r + 8*hi, N = lane%16
#pragma unroll
  for (int tsub = 0; tsub < 4; ++tsub) {
    v8f a = (tsub == 0) ? acc0 : (tsub == 1) ? acc1 : (tsub == 2) ? acc2 : acc3;
    int nCol = (ntBase + tsub) * 16 + l16;
    float bv = bias ? bias[nCol] : 0.f;
#pragma unroll
    for (int r = 0; r < 8; ++r)
      C[(size_t)(mBase + r + hi * 8) * Nc + nCol] = a[r] + bv;
  }
}

// ---------------- edge feature projection: e[E,256] = ea[E,6] @ W[6,256] ----------------
__global__ void k_edge_e(const float* __restrict__ ea, const float* __restrict__ W,
                         float* __restrict__ e) {
  long long t = (long long)blockIdx.x * blockDim.x + threadIdx.x;
  if (t >= (long long)EE * HC) return;
  int eid = (int)(t >> 8), j = (int)(t & 255);
  const float* a = ea + (size_t)eid * 6;
  float v = a[0] * W[j]       + a[1] * W[256 + j]  + a[2] * W[512 + j]
          + a[3] * W[768 + j] + a[4] * W[1024 + j] + a[5] * W[1280 + j];
  __builtin_nontemporal_store(v, &e[t]);     // 205MB stream: NT store policy
}

// ---------------- attention: alpha + segment-max ----------------
__global__ void k_edge_alpha(const float* __restrict__ qkvs, const float* __restrict__ ebuf,
                             const int* __restrict__ src, const int* __restrict__ dst,
                             float* __restrict__ alpha, unsigned* __restrict__ amax) {
  int t = blockIdx.x * blockDim.x + threadIdx.x;
  if (t >= 2 * EE) return;
  int eid = t >> 1, h = t & 1;
  int s = src[eid], d = dst[eid];
  const float4* qp = (const float4*)(qkvs + (size_t)d * NC1 + h * 128);
  const float4* kp = (const float4*)(qkvs + (size_t)s * NC1 + 256 + h * 128);
  const float4* ep = (const float4*)(ebuf + (size_t)eid * HC + h * 128);
  float acc = 0.f;
#pragma unroll 8
  for (int i = 0; i < 32; ++i) {
    float4 q = qp[i], k = kp[i], e = ep[i];
    acc += q.x * (k.x + e.x) + q.y * (k.y + e.y) + q.z * (k.z + e.z) + q.w * (k.w + e.w);
  }
  acc *= SCALE;
  alpha[t] = acc;
  atomicMax(&amax[d * 2 + h], fenc(acc));
}

// ---------------- exp + segment-sum ----------------
__global__ void k_edge_exp(const int* __restrict__ dst, const unsigned* __restrict__ amax,
                           float* __restrict__ alpha, float* __restrict__ den) {
  int t = blockIdx.x * blockDim.x + threadIdx.x;
  if (t >= 2 * EE) return;
  int eid = t >> 1, h = t & 1;
  int d = dst[eid];
  float m = fdec(amax[d * 2 + h]);
  if (!(m > -3.0e38f && m < 3.0e38f)) m = 0.f;
  float ex = __expf(alpha[t] - m);
  alpha[t] = ex;                          // reuse buffer as ex
  atomicAdd(&den[d * 2 + h], ex);
}

// ---------------- weighted message scatter-add ----------------
__global__ void k_edge_msg(const float* __restrict__ qkvs, const float* __restrict__ ebuf,
                           const int* __restrict__ src, const int* __restrict__ dst,
                           const float* __restrict__ ex, const float* __restrict__ den,
                           float* __restrict__ agg) {
  long long t = (long long)blockIdx.x * blockDim.x + threadIdx.x;
  if (t >= (long long)EE * 64) return;
  int eid = (int)(t >> 6);
  int g = (int)(t & 63);
  int h = g >> 5;
  int c = (g & 31) * 4;
  int s = src[eid], d = dst[eid];
  float a = ex[eid * 2 + h] / (den[d * 2 + h] + 1e-16f);
  float4 v = *(const float4*)(qkvs + (size_t)s * NC1 + 512 + h * 128 + c);
  f4v e = __builtin_nontemporal_load((const f4v*)(ebuf + (size_t)eid * HC + h * 128 + c));
  float* o = agg + (size_t)d * HC + h * 128 + c;
  atomicAdd(o + 0, (v.x + e[0]) * a);
  atomicAdd(o + 1, (v.y + e[1]) * a);
  atomicAdd(o + 2, (v.z + e[2]) * a);
  atomicAdd(o + 3, (v.w + e[3]) * a);
}

// ---------------- head-mean + skip, emit bf16 activations ----------------
__global__ void k_node_finish(const float* __restrict__ agg, const float* __restrict__ qkvs,
                              unsigned short* __restrict__ xb) {
  int t = blockIdx.x * blockDim.x + threadIdx.x;
  if (t >= NN * CC) return;
  int n = t >> 7, c = t & 127;
  float val = 0.5f * (agg[(size_t)n * HC + c] + agg[(size_t)n * HC + 128 + c])
            + qkvs[(size_t)n * NC1 + 768 + c];
  xb[t] = f2bf(val);
}

// ---------------- edge predictor: relu(A[src]+B[dst]+bp1) . Wp2 -> sigmoid ----------------
// bp1/Wp2 staged into LDS with async-to-LDS (ASYNCcnt path), consumed via ds_load_b128.
// Grid divides E exactly (200000 / 8 edges-per-block), so barriers are uniform.
__global__ void k_pred_edge(const float* __restrict__ AB, const int* __restrict__ src,
                            const int* __restrict__ dst, const float* __restrict__ bp1,
                            const float* __restrict__ Wp2, const float* __restrict__ bp2,
                            float* __restrict__ out) {
  __shared__ float smem[256];            // [0:128) = bp1, [128:256) = Wp2
  const int tid = threadIdx.x;
  if (tid < 64) {                        // waves 0,1 fully active: EXEC all-ones per wave
    const float* g = (tid < 32) ? (bp1 + tid * 4) : (Wp2 + (tid - 32) * 4);
    unsigned ldsOff = (unsigned)(size_t)(&smem[(tid < 32) ? tid * 4 : 128 + (tid - 32) * 4]);
    asm volatile("global_load_async_to_lds_b128 %0, %1, off"
                 :: "v"(ldsOff), "v"(g) : "memory");
  }
  asm volatile("s_wait_asynccnt 0x0" ::: "memory");
  __syncthreads();

  const int lane = tid & 31;
  const int eid = blockIdx.x * 8 + (tid >> 5);
  const int s = src[eid], d = dst[eid];
  const int c = lane * 4;
  float4 a = *(const float4*)(AB + (size_t)s * 256 + c);
  float4 b = *(const float4*)(AB + (size_t)d * 256 + 128 + c);
  float4 p = *(const float4*)(smem + c);
  float4 w = *(const float4*)(smem + 128 + c);
  float sum = fmaxf(a.x + b.x + p.x, 0.f) * w.x
            + fmaxf(a.y + b.y + p.y, 0.f) * w.y
            + fmaxf(a.z + b.z + p.z, 0.f) * w.z
            + fmaxf(a.w + b.w + p.w, 0.f) * w.w;
#pragma unroll
  for (int m = 16; m >= 1; m >>= 1) sum += __shfl_xor(sum, m, 32);
  if (lane == 0) out[eid] = 1.f / (1.f + __expf(-(sum + bp2[0])));
}

// =====================================================================
static inline int cdiv(int a, int b) { return (a + b - 1) / b; }

extern "C" void kernel_launch(void* const* d_in, const int* in_sizes, int n_in,
                              void* d_out, int out_size, void* d_ws, size_t ws_size,
                              hipStream_t stream) {
  const float* x  = (const float*)d_in[0];
  const float* ea = (const float*)d_in[1];
  const int*   ei = (const int*)d_in[2];
  const int* src = ei;
  const int* dst = ei + EE;
  const float *Wq1 = (const float*)d_in[3],  *bq1 = (const float*)d_in[4];
  const float *Wk1 = (const float*)d_in[5],  *bk1 = (const float*)d_in[6];
  const float *Wv1 = (const float*)d_in[7],  *bv1 = (const float*)d_in[8];
  const float *We1 = (const float*)d_in[9];
  const float *Ws1 = (const float*)d_in[10], *bs1 = (const float*)d_in[11];
  const float *Wq2 = (const float*)d_in[12], *bq2 = (const float*)d_in[13];
  const float *Wk2 = (const float*)d_in[14], *bk2 = (const float*)d_in[15];
  const float *Wv2 = (const float*)d_in[16], *bv2 = (const float*)d_in[17];
  const float *We2 = (const float*)d_in[18];
  const float *Ws2 = (const float*)d_in[19], *bs2 = (const float*)d_in[20];
  const float *Wp1 = (const float*)d_in[21], *bp1 = (const float*)d_in[22];
  const float *Wp2 = (const float*)d_in[23], *bp2 = (const float*)d_in[24];
  float* out = (float*)d_out;
  (void)in_sizes; (void)n_in; (void)out_size; (void)ws_size;

  // ---- workspace carve-up (~267 MB) ----
  char* ws = (char*)d_ws;
  size_t off = 0;
  auto alloc = [&](size_t bytes) { size_t o = off; off = (off + bytes + 255) & ~(size_t)255; return o; };
  unsigned short* xb     = (unsigned short*)(ws + alloc((size_t)NN * F_IN * 2));
  unsigned short* w1cat  = (unsigned short*)(ws + alloc((size_t)F_IN * NC1 * 2));
  float*          b1cat  = (float*)         (ws + alloc((size_t)NC1 * 4));
  unsigned short* w2cat  = (unsigned short*)(ws + alloc((size_t)CC * NC1 * 2));
  float*          b2cat  = (float*)         (ws + alloc((size_t)NC1 * 4));
  unsigned short* wpcat  = (unsigned short*)(ws + alloc((size_t)CC * 256 * 2));
  float*          we2eff = (float*)         (ws + alloc((size_t)6 * 256 * 4));
  float*          qkvs   = (float*)         (ws + alloc((size_t)NN * NC1 * 4));
  float*          ebuf   = (float*)         (ws + alloc((size_t)EE * HC * 4));
  float*          alpha  = (float*)         (ws + alloc((size_t)EE * 2 * 4));
  unsigned*       amax   = (unsigned*)      (ws + alloc((size_t)NN * 2 * 4));
  float*          den    = (float*)         (ws + alloc((size_t)NN * 2 * 4));
  float*          agg    = (float*)         (ws + alloc((size_t)NN * HC * 4));  // reused as AB
  unsigned short* xcb    = (unsigned short*)(ws + alloc((size_t)NN * CC * 2));

  const int B = 256;
  const unsigned encNegInf = 0x007FFFFFu;  // fenc(-inf)

  // ---- one-time packs / converts ----
  k_cvt_bf16 <<<cdiv(NN * F_IN, B), B, 0, stream>>>(x, xb, NN * F_IN);
  k_pack_wcat_swz<<<cdiv(F_IN * NC1, B), B, 0, stream>>>(Wq1, Wk1, Wv1, Ws1, w1cat, F_IN);
  k_pack_bcat<<<cdiv(NC1, B), B, 0, stream>>>(bq1, bk1, bv1, bs1, b1cat);
  k_pack_wcat_swz<<<cdiv(CC * NC1, B), B, 0, stream>>>(Wq2, Wk2, Wv2, Ws2, w2cat, CC);
  k_pack_bcat<<<cdiv(NC1, B), B, 0, stream>>>(bq2, bk2, bv2, bs2, b2cat);
  k_pack_wpcat_swz<<<cdiv(CC * 256, B), B, 0, stream>>>(Wp1, wpcat);
  k_we2eff<<<cdiv(6 * 256, B), B, 0, stream>>>(We1, We2, we2eff);

  // ================= layer 1 =================
  k_gemm_bf16<<<dim3(NC1 / 64, cdiv(NN, 64)), 128, 0, stream>>>(xb, w1cat, b1cat, qkvs, NN, F_IN, NC1);
  k_edge_e<<<cdiv(EE * HC, B), B, 0, stream>>>(ea, We1, ebuf);
  k_fill_u32<<<cdiv(NN * 2, B), B, 0, stream>>>(amax, encNegInf, NN * 2);
  k_fill_f32<<<cdiv(NN * 2, B), B, 0, stream>>>(den, 0.f, NN * 2);
  k_fill_f32<<<cdiv(NN * HC, B), B, 0, stream>>>(agg, 0.f, NN * HC);
  k_edge_alpha<<<cdiv(2 * EE, B), B, 0, stream>>>(qkvs, ebuf, src, dst, alpha, amax);
  k_edge_exp  <<<cdiv(2 * EE, B), B, 0, stream>>>(dst, amax, alpha, den);
  k_edge_msg  <<<cdiv(EE * 64, B), B, 0, stream>>>(qkvs, ebuf, src, dst, alpha, den, agg);
  k_node_finish<<<cdiv(NN * CC, B), B, 0, stream>>>(agg, qkvs, xcb);   // -> x2 (bf16)

  // ================= layer 2 =================
  k_gemm_bf16<<<dim3(NC1 / 64, cdiv(NN, 64)), 128, 0, stream>>>(xcb, w2cat, b2cat, qkvs, NN, CC, NC1);
  k_edge_e<<<cdiv(EE * HC, B), B, 0, stream>>>(ea, we2eff, ebuf);      // e2 = ea @ (mean-head(We1) @ We2)
  k_fill_u32<<<cdiv(NN * 2, B), B, 0, stream>>>(amax, encNegInf, NN * 2);
  k_fill_f32<<<cdiv(NN * 2, B), B, 0, stream>>>(den, 0.f, NN * 2);
  k_fill_f32<<<cdiv(NN * HC, B), B, 0, stream>>>(agg, 0.f, NN * HC);
  k_edge_alpha<<<cdiv(2 * EE, B), B, 0, stream>>>(qkvs, ebuf, src, dst, alpha, amax);
  k_edge_exp  <<<cdiv(2 * EE, B), B, 0, stream>>>(dst, amax, alpha, den);
  k_edge_msg  <<<cdiv(EE * 64, B), B, 0, stream>>>(qkvs, ebuf, src, dst, alpha, den, agg);
  k_node_finish<<<cdiv(NN * CC, B), B, 0, stream>>>(agg, qkvs, xcb);   // -> x3 (bf16)

  // ================= edge predictor =================
  k_gemm_bf16<<<dim3(256 / 64, cdiv(NN, 64)), 128, 0, stream>>>(xcb, wpcat, nullptr, agg, NN, CC, 256);
  k_pred_edge<<<EE / 8, 256, 0, stream>>>(agg, src, dst, bp1, Wp2, bp2, out);
}
